// MultichannelMultiheadAttention_73186242724558
// MI455X (gfx1250) — matrix operand order repaired and checked by reading, compiled, not run on hardware
//
#include <hip/hip_runtime.h>
#include <hip/hip_bf16.h>

// MI455X / gfx1250. All GEMM-shaped work through v_wmma_f32_16x16x32_bf16
// (wave32, f32 accumulate). Operands are pre-packed to bf16 in WMMA-friendly
// orientations, and every operand fetch goes through 16B-aligned vector loads
// so the hot loops are pure {s_clause + global_load_b128 -> v_wmma}:
//   - weights   : bf16 [M,K] row-major  (A-tiles: 2x16B aligned runs per lane)
//   - activ.    : bf16 [N,K] transposed (B-tiles: 2x16B aligned runs per lane)
//   - V         : bf16 [head,d,w]       (PV B-tiles contiguous)
//   - attn out  : bf16 [c,w,F]          (feeds o-proj GEMM directly)

typedef __bf16 bf16_t;
typedef __attribute__((ext_vector_type(16))) __bf16 v16bf;
typedef __attribute__((ext_vector_type(8)))  __bf16 v8bf;   // 16B, align 16
typedef __attribute__((ext_vector_type(8)))  float  v8f;

#define C_DIM 8
#define NH    8
#define F_DIM 256
#define W_DIM 1024
#define D_DIM 32

__device__ __forceinline__ bf16_t f2bf(float x) { return (bf16_t)x; }

// Two 16B-aligned 8xbf16 loads glued into one WMMA operand register block.
__device__ __forceinline__ v16bf load2x8(const bf16_t* p, int off_lo, int off_hi) {
  const v8bf lo = *(const v8bf*)(p + off_lo);
  const v8bf hi = *(const v8bf*)(p + off_hi);
  return __builtin_shufflevector(lo, hi, 0, 1, 2, 3, 4, 5, 6, 7,
                                         8, 9, 10, 11, 12, 13, 14, 15);
}

// ---------------------------------------------------------------------------
// Elementwise f32 -> bf16 pack (weights).
// ---------------------------------------------------------------------------
__global__ void pack_bf16_kernel(const float* __restrict__ src,
                                 bf16_t* __restrict__ dst, int n)
{
  const int idx = blockIdx.x * blockDim.x + threadIdx.x;
  if (idx < n) dst[idx] = f2bf(src[idx]);
}

// ---------------------------------------------------------------------------
// x [C,256,1024] f32 -> XT [C,1024,256] bf16 (activations in [N,K] order).
// ---------------------------------------------------------------------------
__global__ void pack_xT_kernel(const float* __restrict__ X,
                               bf16_t* __restrict__ XT)
{
  const int idx = blockIdx.x * blockDim.x + threadIdx.x;
  if (idx >= C_DIM * F_DIM * W_DIM) return;
  const int f = idx & (F_DIM - 1);
  const int w = (idx >> 8) & (W_DIM - 1);
  const int c = idx >> 18;
  XT[idx] = f2bf(X[((size_t)(c * F_DIM + f)) * W_DIM + w]);
}

// ---------------------------------------------------------------------------
// Per-channel GEMM, all-bf16 operands: Y[c] = W[c](256x256) * X[c](256x1024).
// One wave computes 16(M) x 64(N); K-loop of 8 x 32.
// ---------------------------------------------------------------------------
__global__ __launch_bounds__(32) void mc_gemm_bf16_kernel(
    const bf16_t* __restrict__ Wbf,  // [C,256,256]  (M,K)
    const bf16_t* __restrict__ XT,   // [C,1024,256] (N,K)
    const float* __restrict__ bpw,   // [C,256] or nullptr
    float* __restrict__ Y)           // [C,256,1024]
{
  const int c    = blockIdx.z;
  const int m0   = blockIdx.y * 16;
  const int n0   = blockIdx.x * 64;
  const int lane = threadIdx.x;

  // wave32 WMMA operand mapping used here (contiguous k; A and B agree)
  const int mA = lane & 15;           // A: lane -> M row
  const int kA = (lane >> 4) * 16;    // A: j -> K=kA+j
  const int nB = lane & 15;           // B: lane -> N col
  const int kB = (lane >> 4) * 16;    // B: j -> K=kB+j

  const bf16_t* Wc = Wbf + (size_t)c * F_DIM * F_DIM;
  const bf16_t* Xc = XT  + (size_t)c * W_DIM * F_DIM;

  v8f acc0 = {}, acc1 = {}, acc2 = {}, acc3 = {};

  for (int k0 = 0; k0 < F_DIM; k0 += 32) {
    const bf16_t* wrow = Wc + (size_t)(m0 + mA) * F_DIM + k0;   // 16B aligned +kA
    const v16bf a = load2x8(wrow, kA, kA + 8);

    const bf16_t* x0r = Xc + (size_t)(n0 + nB) * F_DIM + k0 + kB;  // 32B aligned
    const v16bf b0 = load2x8(x0r, 0, 8);
    const v16bf b1 = load2x8(x0r + (size_t)16 * F_DIM, 0, 8);
    const v16bf b2 = load2x8(x0r + (size_t)32 * F_DIM, 0, 8);
    const v16bf b3 = load2x8(x0r + (size_t)48 * F_DIM, 0, 8);

    acc0 = __builtin_amdgcn_wmma_f32_16x16x32_bf16(false, a, false, b0, (short)0, acc0, false, false);
    acc1 = __builtin_amdgcn_wmma_f32_16x16x32_bf16(false, a, false, b1, (short)0, acc1, false, false);
    acc2 = __builtin_amdgcn_wmma_f32_16x16x32_bf16(false, a, false, b2, (short)0, acc2, false, false);
    acc3 = __builtin_amdgcn_wmma_f32_16x16x32_bf16(false, a, false, b3, (short)0, acc3, false, false);
  }

  const int mrow = (lane >> 4) * 8;   // C/D: reg r -> row r + 8*(lane>=16)
#pragma unroll
  for (int r = 0; r < 8; ++r) {
    const int m = m0 + mrow + r;
    const float bias = bpw ? bpw[c * F_DIM + m] : 0.0f;
    float* yrow = Y + ((size_t)(c * F_DIM + m)) * W_DIM + n0 + nB;
    yrow[0]  = acc0[r] + bias;
    yrow[16] = acc1[r] + bias;
    yrow[32] = acc2[r] + bias;
    yrow[48] = acc3[r] + bias;
  }
}

// ---------------------------------------------------------------------------
// Cross-channel conv (1,3), padding (0,1).
// ---------------------------------------------------------------------------
__global__ void conv_w3_kernel(const float* __restrict__ Y,
                               const float* __restrict__ Wc3, // [C,C,1,3]
                               const float* __restrict__ bc,  // [C]
                               float* __restrict__ Z)
{
  const int idx = blockIdx.x * blockDim.x + threadIdx.x;
  if (idx >= C_DIM * F_DIM * W_DIM) return;
  const int w = idx & (W_DIM - 1);
  const int f = (idx >> 10) & (F_DIM - 1);
  const int o = idx >> 18;
  float acc = bc[o];
#pragma unroll
  for (int i = 0; i < C_DIM; ++i) {
    const float* yrow = Y + ((size_t)(i * F_DIM + f)) * W_DIM;
    const float* wk   = Wc3 + (o * C_DIM + i) * 3;
    const float xm = (w > 0)         ? yrow[w - 1] : 0.0f;
    const float x0 = yrow[w];
    const float xp = (w < W_DIM - 1) ? yrow[w + 1] : 0.0f;
    acc = fmaf(xm, wk[0], fmaf(x0, wk[1], fmaf(xp, wk[2], acc)));
  }
  Z[idx] = acc;
}

// ---------------------------------------------------------------------------
// Head packing with optional interleaved rotary.
//   transpose_v == 0 : H[head, w, d]   (Q / K score operands)
//   transpose_v == 1 : H[head, d, w]   (V, so PV B-tiles are contiguous)
// ---------------------------------------------------------------------------
__global__ void head_pack_kernel(const float* __restrict__ Z,
                                 bf16_t* __restrict__ H,
                                 int do_rot, int transpose_v)
{
  const int idx = blockIdx.x * blockDim.x + threadIdx.x;
  if (idx >= C_DIM * NH * W_DIM * (D_DIM / 2)) return;
  const int i    = idx & 15;                 // rotary pair index
  const int w    = (idx >> 4) & (W_DIM - 1);
  const int head = idx >> 14;
  const int c    = head >> 3;
  const int nh   = head & 7;

  const float* base = Z + ((size_t)(c * F_DIM + nh * D_DIM)) * W_DIM + w;
  const float x0 = base[(size_t)(2 * i) * W_DIM];
  const float x1 = base[(size_t)(2 * i + 1) * W_DIM];
  float o0 = x0, o1 = x1;
  if (do_rot) {
    const float inv_freq = __powf(10000.0f, -((float)(2 * i)) / (float)D_DIM);
    const float ang = (float)w * inv_freq;
    float sn, cs;
    __sincosf(ang, &sn, &cs);
    o0 = x0 * cs - x1 * sn;
    o1 = x1 * cs + x0 * sn;
  }
  if (transpose_v) {
    bf16_t* hb = H + ((size_t)head * D_DIM) * W_DIM + w;
    hb[(size_t)(2 * i) * W_DIM]     = f2bf(o0);
    hb[(size_t)(2 * i + 1) * W_DIM] = f2bf(o1);
  } else {
    bf16_t* hb = H + ((size_t)head * W_DIM + w) * D_DIM;
    hb[2 * i]     = f2bf(o0);
    hb[2 * i + 1] = f2bf(o1);
  }
}

// ---------------------------------------------------------------------------
// Flash attention, one wave per (head, 16-row q-tile). Online softmax over
// 32-key chunks: 2 WMMA scores + 2 WMMA PV per chunk. P transposed
// C-layout -> A-layout via a 1KB LDS bounce (aligned ds_load_b128 reads).
// Output written bf16 in [c,w,F] (the o-projection's B-operand layout).
// ---------------------------------------------------------------------------
__global__ __launch_bounds__(32) void flash_attn_kernel(
    const bf16_t* __restrict__ Q,   // [head, w, d]
    const bf16_t* __restrict__ K,   // [head, w, d]
    const bf16_t* __restrict__ V,   // [head, d, w]  (transposed)
    bf16_t* __restrict__ AT)        // [C, 1024, 256]
{
  __shared__ alignas(16) bf16_t sP[16 * 32];

  const int q0   = blockIdx.x * 16;
  const int head = blockIdx.y;
  const int c    = head >> 3;
  const int nh   = head & 7;
  const int lane = threadIdx.x;
  const size_t hoff = (size_t)head * W_DIM * D_DIM;

  const int mA = lane & 15;
  const int kA = (lane >> 4) * 16;
  const int nB = lane & 15;
  const int kB = (lane >> 4) * 16;

  // Q tile in A layout (M = q row, K = d); rows are 64B aligned
  const v16bf aq = load2x8(Q + hoff + (size_t)(q0 + mA) * D_DIM, kA, kA + 8);

  const float scale = 0.0625f;   // 1/sqrt(F=256)
  float mrun[8], lrun[8];
#pragma unroll
  for (int r = 0; r < 8; ++r) { mrun[r] = -1e30f; lrun[r] = 0.0f; }
  v8f o0 = {}, o1 = {};

  for (int kb = 0; kb < W_DIM; kb += 32) {
    // --- scores: Q(16x32) x K^T(32x16) twice; 2x16B aligned runs per lane ---
    const bf16_t* kbase = K + hoff + (size_t)(kb + nB) * D_DIM + kB;
    __builtin_prefetch(kbase + 32 * D_DIM, 0, 1);
    const v16bf bk0 = load2x8(kbase, 0, 8);
    const v16bf bk1 = load2x8(kbase + 16 * D_DIM, 0, 8);
    v8f s0 = {}, s1 = {};
    s0 = __builtin_amdgcn_wmma_f32_16x16x32_bf16(false, aq, false, bk0, (short)0, s0, false, false);
    s1 = __builtin_amdgcn_wmma_f32_16x16x32_bf16(false, aq, false, bk1, (short)0, s1, false, false);

    // --- online softmax; C-layout rows live in 16-lane half-groups ---
#pragma unroll
    for (int r = 0; r < 8; ++r) {
      const float v0 = s0[r] * scale;
      const float v1 = s1[r] * scale;
      float vm = fmaxf(v0, v1);
      vm = fmaxf(vm, __shfl_xor(vm, 1));
      vm = fmaxf(vm, __shfl_xor(vm, 2));
      vm = fmaxf(vm, __shfl_xor(vm, 4));
      vm = fmaxf(vm, __shfl_xor(vm, 8));
      const float mnew = fmaxf(mrun[r], vm);
      const float p0 = __expf(v0 - mnew);
      const float p1 = __expf(v1 - mnew);
      float rs = p0 + p1;
      rs += __shfl_xor(rs, 1);
      rs += __shfl_xor(rs, 2);
      rs += __shfl_xor(rs, 4);
      rs += __shfl_xor(rs, 8);
      const float corr = __expf(mrun[r] - mnew);
      lrun[r] = lrun[r] * corr + rs;
      mrun[r] = mnew;
      o0[r] *= corr;
      o1[r] *= corr;
      const int m = r + ((lane >> 4) << 3);
      sP[m * 32 + (lane & 15)]      = f2bf(p0);
      sP[m * 32 + 16 + (lane & 15)] = f2bf(p1);
    }
    asm volatile("s_wait_dscnt 0" ::: "memory");   // LDS stores visible to reads

    // P in A layout (M = q row, K = key-in-chunk); 16B-aligned ds reads
    const v16bf pa = load2x8(sP + mA * 32, kA, kA + 8);

    // --- PV: P(16x32) x V(32x32); V transposed so runs are contiguous ---
    const bf16_t* vb0 = V + ((size_t)head * D_DIM + nB) * W_DIM + kb + kB;
    __builtin_prefetch(vb0 + 32, 0, 1);
    const v16bf bv0 = load2x8(vb0, 0, 8);
    const v16bf bv1 = load2x8(vb0 + (size_t)16 * W_DIM, 0, 8);
    o0 = __builtin_amdgcn_wmma_f32_16x16x32_bf16(false, pa, false, bv0, (short)0, o0, false, false);
    o1 = __builtin_amdgcn_wmma_f32_16x16x32_bf16(false, pa, false, bv1, (short)0, o1, false, false);
  }

  // epilogue: normalize, emit bf16 in [c, w, F] (o-proj B operand layout)
#pragma unroll
  for (int r = 0; r < 8; ++r) {
    const int m = q0 + r + ((lane >> 4) << 3);
    const float inv_l = 1.0f / lrun[r];
    const int d = lane & 15;
    bf16_t* arow = AT + ((size_t)c * W_DIM + m) * F_DIM + nh * D_DIM;
    arow[d]      = f2bf(o0[r] * inv_l);
    arow[16 + d] = f2bf(o1[r] * inv_l);
  }
}

// ---------------------------------------------------------------------------
// Final 8x8 channel mix: out[o,f,w] = sum_c O1[c,f,w] * Wdw[o,c]
// ---------------------------------------------------------------------------
__global__ void chmix_kernel(const float* __restrict__ O1,
                             const float* __restrict__ Wdw,
                             float* __restrict__ out)
{
  const int idx = blockIdx.x * blockDim.x + threadIdx.x;
  if (idx >= C_DIM * F_DIM * W_DIM) return;
  const int fw = idx & (F_DIM * W_DIM - 1);
  const int o  = idx >> 18;
  float acc = 0.0f;
#pragma unroll
  for (int c = 0; c < C_DIM; ++c)
    acc = fmaf(O1[(size_t)c * F_DIM * W_DIM + fw], Wdw[o * C_DIM + c], acc);
  out[idx] = acc;
}

// ---------------------------------------------------------------------------
extern "C" void kernel_launch(void* const* d_in, const int* in_sizes, int n_in,
                              void* d_out, int out_size, void* d_ws, size_t ws_size,
                              hipStream_t stream) {
  const float* x = (const float*)d_in[0];
  // q: 1..4, k: 5..8, v: 9..12
  const float* Wpw[3]  = { (const float*)d_in[1], (const float*)d_in[5],  (const float*)d_in[9]  };
  const float* bpw[3]  = { (const float*)d_in[2], (const float*)d_in[6],  (const float*)d_in[10] };
  const float* Wcv[3]  = { (const float*)d_in[3], (const float*)d_in[7],  (const float*)d_in[11] };
  const float* bcv[3]  = { (const float*)d_in[4], (const float*)d_in[8],  (const float*)d_in[12] };
  const float* Wo_pw   = (const float*)d_in[13];
  const float* Wo_dw   = (const float*)d_in[14];

  const size_t MB = 1024 * 1024;
  char* ws = (char*)d_ws;
  float*  Yf   = (float*)ws;                         //  8 MB f32 scratch (pw out)
  float*  Zf   = (float*)(ws + 8  * MB);             //  8 MB f32 scratch (conv / o-proj out)
  bf16_t* HBF[3] = {
    (bf16_t*)(ws + 16 * MB),                         // Q bf16 [head,w,d], 4 MB
    (bf16_t*)(ws + 20 * MB),                         // K bf16 [head,w,d], 4 MB
    (bf16_t*)(ws + 24 * MB),                         // V bf16 [head,d,w], 4 MB
  };
  bf16_t* XTbf = (bf16_t*)(ws + 28 * MB);            // x  bf16 [C,1024,256], 4 MB
  bf16_t* ATbf = (bf16_t*)(ws + 32 * MB);            // attn out bf16 [C,1024,256], 4 MB
  bf16_t* Wbf[4] = {                                 // weights bf16 [C,256,256], 1 MB each
    (bf16_t*)(ws + 36 * MB), (bf16_t*)(ws + 37 * MB),
    (bf16_t*)(ws + 38 * MB), (bf16_t*)(ws + 39 * MB),
  };

  const dim3 gemm_grid(W_DIM / 64, F_DIM / 16, C_DIM);   // wave-per-16x64-tile
  const int  n_elem = C_DIM * F_DIM * W_DIM;             // 2M
  const int  n_wamt = C_DIM * F_DIM * F_DIM;             // 512K per weight tensor
  const int  n_pack = C_DIM * NH * W_DIM * (D_DIM / 2);  // 1M

  // operand packing (bf16, GEMM-friendly orientations)
  pack_xT_kernel<<<(n_elem + 255) / 256, 256, 0, stream>>>(x, XTbf);
  pack_bf16_kernel<<<(n_wamt + 255) / 256, 256, 0, stream>>>(Wpw[0], Wbf[0], n_wamt);
  pack_bf16_kernel<<<(n_wamt + 255) / 256, 256, 0, stream>>>(Wpw[1], Wbf[1], n_wamt);
  pack_bf16_kernel<<<(n_wamt + 255) / 256, 256, 0, stream>>>(Wpw[2], Wbf[2], n_wamt);
  pack_bf16_kernel<<<(n_wamt + 255) / 256, 256, 0, stream>>>(Wo_pw,  Wbf[3], n_wamt);

  // q/k/v projection -> conv -> rotary/head pack
  for (int p = 0; p < 3; ++p) {
    mc_gemm_bf16_kernel<<<gemm_grid, 32, 0, stream>>>(Wbf[p], XTbf, bpw[p], Yf);
    conv_w3_kernel<<<(n_elem + 255) / 256, 256, 0, stream>>>(Yf, Wcv[p], bcv[p], Zf);
    head_pack_kernel<<<(n_pack + 255) / 256, 256, 0, stream>>>(
        Zf, HBF[p], (p < 2) ? 1 : 0, (p == 2) ? 1 : 0);
  }

  // attention (writes bf16 [c,w,F] directly)
  flash_attn_kernel<<<dim3(W_DIM / 16, C_DIM * NH), 32, 0, stream>>>(
      HBF[0], HBF[1], HBF[2], ATbf);

  // output projection + channel mix
  mc_gemm_bf16_kernel<<<gemm_grid, 32, 0, stream>>>(Wbf[3], ATbf, nullptr, Zf);
  chmix_kernel<<<(n_elem + 255) / 256, 256, 0, stream>>>(Zf, Wo_dw, (float*)d_out);
}
